// DualGRUDecoder_65816078844065
// MI455X (gfx1250) — compile-verified, matrix operand order checked by reading, not run on hardware
//
#include <hip/hip_runtime.h>
#include <hip/hip_bf16.h>

// ---------------------------------------------------------------------------
// Problem constants (from reference)
// ---------------------------------------------------------------------------
constexpr int V   = 50000;
constexpr int EMB = 256;
constexpr int DEC = 512;
constexpr int E2  = 1024;
constexpr int Bn  = 128;
constexpr int S1  = 200;
constexpr int S2  = 200;
constexpr float NEGV = -1e10f;

typedef __attribute__((ext_vector_type(16))) __bf16 v16bf;
typedef __attribute__((ext_vector_type(8)))  float  v8f;

// ---------------------------------------------------------------------------
// Generic WMMA GEMM:  C[M,N] = A[M,K] (f32 row-major) * W[N,K]^T  (+ bias[N])
// One wave per 16x(16*NT) output strip: the A-fragment is loaded once per
// k-step and reused for NT WMMAs (cuts A-side L2 traffic by NT).
// grid.x = N/(16*NT), grid.y = M/16, block = 32.
// A-frag layout (16-bit A 16x32, ISA 7.12.2): lane<16 -> K {0..7,16..23},
// lane>=16 -> K {8..15,24..31}; M = lane&15. B mirrored with N = lane&15.
// C/D: VGPR j -> M = j + 8*(lane>=16), N = lane&15.
// ---------------------------------------------------------------------------
template <typename WT, int NT>
__global__ __launch_bounds__(32)
void wmma_gemm_bt(const float* __restrict__ A, int lda,
                  const WT* __restrict__ W, int ldw,
                  const float* __restrict__ bias,
                  float* __restrict__ C, int ldc, int K) {
  const int l      = threadIdx.x;
  const int lane15 = l & 15;
  const int half   = l >> 4;
  const int kb     = half * 8;
  const int n0     = blockIdx.x * (16 * NT);
  const int m0     = blockIdx.y * 16;

  const float* ap = A + (size_t)(m0 + lane15) * lda + kb;

  v8f acc[NT];
#pragma unroll
  for (int t = 0; t < NT; ++t) { v8f z = {}; acc[t] = z; }

  for (int k0 = 0; k0 < K; k0 += 32) {
    v16bf a;
#pragma unroll
    for (int i = 0; i < 8; ++i) {
      a[i]     = (__bf16)ap[k0 + i];
      a[i + 8] = (__bf16)ap[k0 + 16 + i];
    }
#pragma unroll
    for (int t = 0; t < NT; ++t) {
      const WT* wp = W + (size_t)(n0 + t * 16 + lane15) * ldw + k0 + kb;
      v16bf b;
#pragma unroll
      for (int i = 0; i < 8; ++i) {
        b[i]     = (__bf16)(float)wp[i];
        b[i + 8] = (__bf16)(float)wp[16 + i];
      }
      acc[t] = __builtin_amdgcn_wmma_f32_16x16x32_bf16(false, a, false, b,
                                                       (short)0, acc[t], false, false);
    }
  }

#pragma unroll
  for (int t = 0; t < NT; ++t) {
    const int col = n0 + t * 16 + lane15;
    const float bv = bias ? bias[col] : 0.0f;
#pragma unroll
    for (int j = 0; j < 8; ++j) {
      const int row = m0 + j + half * 8;
      C[(size_t)row * ldc + col] = acc[t][j] + bv;
    }
  }
}

// ---------------------------------------------------------------------------
// Fused Bahdanau energy + reduce:
//   scores[s,b] = sum_n v[n] * tanh( hp[b,n] + sum_k enc[s,b,k]*Wt[n,k] )
// enc: [S,Bn,E2] f32, Wt: [512,1024] bf16 ([N,K]), hp: [Bn,512] (= h@W_top+b)
// Two s-values per workgroup: every B-fragment (Wt) load feeds two WMMAs,
// halving the dominant W-side L2 traffic.
// grid.x = Bn/16, grid.y = S/2, block = 128 (4 waves, 8 N-tiles each).
// ---------------------------------------------------------------------------
__global__ __launch_bounds__(128)
void attn_scores(const float* __restrict__ enc, const __bf16* __restrict__ Wt,
                 const float* __restrict__ hp, const float* __restrict__ v,
                 float* __restrict__ scores) {
  __shared__ float rowAcc[2][16];
  const int tid    = threadIdx.x;
  const int wave   = tid >> 5;
  const int l      = tid & 31;
  const int lane15 = l & 15;
  const int half   = l >> 4;
  const int kb     = half * 8;
  const int s0     = blockIdx.y * 2;
  const int b0     = blockIdx.x * 16;

  if (tid < 32) rowAcc[tid >> 4][tid & 15] = 0.0f;
  __syncthreads();

  const float* ap0 = enc + ((size_t)s0 * Bn + b0 + lane15) * E2 + kb;
  const float* ap1 = ap0 + (size_t)Bn * E2;
  const int nbase = wave * 128;

  v8f acc[2][8];
#pragma unroll
  for (int ss = 0; ss < 2; ++ss)
#pragma unroll
    for (int t = 0; t < 8; ++t) { v8f z = {}; acc[ss][t] = z; }

  for (int k0 = 0; k0 < E2; k0 += 32) {
    v16bf a0, a1;
#pragma unroll
    for (int i = 0; i < 8; ++i) {
      a0[i]     = (__bf16)ap0[k0 + i];
      a0[i + 8] = (__bf16)ap0[k0 + 16 + i];
      a1[i]     = (__bf16)ap1[k0 + i];
      a1[i + 8] = (__bf16)ap1[k0 + 16 + i];
    }
#pragma unroll
    for (int t = 0; t < 8; ++t) {
      const __bf16* wp = Wt + (size_t)(nbase + t * 16 + lane15) * E2 + k0 + kb;
      v16bf b;
#pragma unroll
      for (int i = 0; i < 8; ++i) { b[i] = wp[i]; b[i + 8] = wp[16 + i]; }
      acc[0][t] = __builtin_amdgcn_wmma_f32_16x16x32_bf16(false, a0, false, b,
                                                          (short)0, acc[0][t], false, false);
      acc[1][t] = __builtin_amdgcn_wmma_f32_16x16x32_bf16(false, a1, false, b,
                                                          (short)0, acc[1][t], false, false);
    }
  }

#pragma unroll
  for (int ss = 0; ss < 2; ++ss) {
#pragma unroll
    for (int t = 0; t < 8; ++t) {
      const int ncol = nbase + t * 16 + lane15;
      const float vn = v[ncol];
#pragma unroll
      for (int j = 0; j < 8; ++j) {
        const int m = j + half * 8;
        const int b = b0 + m;
        float x = tanhf(acc[ss][t][j] + hp[(size_t)b * DEC + ncol]) * vn;
#pragma unroll
        for (int off = 1; off < 16; off <<= 1) x += __shfl_xor(x, off, 32);
        if (lane15 == 0) atomicAdd(&rowAcc[ss][m], x);
      }
    }
  }
  __syncthreads();
  if (tid < 32)
    scores[(size_t)(s0 + (tid >> 4)) * Bn + b0 + (tid & 15)] = rowAcc[tid >> 4][tid & 15];
}

// ---------------------------------------------------------------------------
// Helpers / elementwise kernels
// ---------------------------------------------------------------------------
__global__ void transpose_bf16(const float* __restrict__ W, __bf16* __restrict__ Wt,
                               int Kd, int Nd) {
  const int stride = gridDim.x * blockDim.x;
  const int total = Kd * Nd;
  for (int i = blockIdx.x * blockDim.x + threadIdx.x; i < total; i += stride) {
    const int n = i / Kd, k = i % Kd;
    Wt[(size_t)n * Kd + k] = (__bf16)W[(size_t)k * Nd + n];
  }
}

__global__ void embed_gather(const int* __restrict__ in,
                             const float* __restrict__ table,
                             float* __restrict__ out) {
  const int b = blockIdx.x, e = threadIdx.x;
  out[(size_t)b * EMB + e] = table[(size_t)in[b] * EMB + e];
}

// mode 0: mask = concat(mA[B,SA], mB[B,SB]); mode 1: mask = mA*mB, S = SA
__global__ __launch_bounds__(256)
void masked_softmax(const float* __restrict__ sc, float* __restrict__ out,
                    const int* __restrict__ mA, const int* __restrict__ mB,
                    int SA, int SB, int mode) {
  const int b = blockIdx.x, tid = threadIdx.x;
  const int S = (mode == 0) ? (SA + SB) : SA;
  __shared__ float red[256];

  auto getv = [&](int s) -> float {
    int mbit;
    if (mode == 0) mbit = (s < SA) ? mA[b * SA + s] : mB[b * SB + (s - SA)];
    else           mbit = mA[b * SA + s] * mB[b * SA + s];
    return mbit ? sc[(size_t)s * Bn + b] : NEGV;
  };

  float mx = -3.0e38f;
  for (int s = tid; s < S; s += 256) mx = fmaxf(mx, getv(s));
  red[tid] = mx; __syncthreads();
  for (int o = 128; o > 0; o >>= 1) { if (tid < o) red[tid] = fmaxf(red[tid], red[tid + o]); __syncthreads(); }
  mx = red[0]; __syncthreads();

  float sum = 0.0f;
  for (int s = tid; s < S; s += 256) sum += expf(getv(s) - mx);
  red[tid] = sum; __syncthreads();
  for (int o = 128; o > 0; o >>= 1) { if (tid < o) red[tid] += red[tid + o]; __syncthreads(); }
  const float inv = 1.0f / red[0]; __syncthreads();

  for (int s = tid; s < S; s += 256)
    out[(size_t)s * Bn + b] = expf(getv(s) - mx) * inv;
}

__global__ __launch_bounds__(256)
void weighted_sum(const float* __restrict__ a, const float* __restrict__ e1,
                  const float* __restrict__ e2, float* __restrict__ w) {
  const int e = blockIdx.x * blockDim.x + threadIdx.x;
  const int b = blockIdx.y;
  float acc = 0.0f;
  for (int s = 0; s < S1; ++s) acc += a[s * Bn + b] * e1[((size_t)s * Bn + b) * E2 + e];
  for (int s = 0; s < S2; ++s) acc += a[(S1 + s) * Bn + b] * e2[((size_t)s * Bn + b) * E2 + e];
  w[(size_t)b * E2 + e] = acc;
}

__global__ void concat2(const float* __restrict__ x1, int w1,
                        const float* __restrict__ x2, int w2,
                        float* __restrict__ o, int n) {
  const int stride = gridDim.x * blockDim.x;
  const int wt = w1 + w2;
  for (int i = blockIdx.x * blockDim.x + threadIdx.x; i < n; i += stride) {
    const int b = i / wt, c = i % wt;
    o[i] = (c < w1) ? x1[(size_t)b * w1 + c] : x2[(size_t)b * w2 + (c - w1)];
  }
}

__global__ __launch_bounds__(256)
void gate_kernel(const float* __restrict__ wsum, const float* __restrict__ emb,
                 const float* __restrict__ gw, const float* __restrict__ gb,
                 float* __restrict__ pg) {
  const int b = blockIdx.x, tid = threadIdx.x;
  __shared__ float red[256];
  float acc = 0.0f;
  for (int k = tid; k < E2; k += 256)  acc += wsum[(size_t)b * E2 + k] * gw[k];
  for (int k = tid; k < EMB; k += 256) acc += emb[(size_t)b * EMB + k] * gw[E2 + k];
  red[tid] = acc; __syncthreads();
  for (int o = 128; o > 0; o >>= 1) { if (tid < o) red[tid] += red[tid + o]; __syncthreads(); }
  if (tid == 0) pg[b] = 1.0f / (1.0f + expf(-(red[0] + gb[0])));
}

__global__ void gru_elem(const float* __restrict__ gx, const float* __restrict__ gh,
                         const float* __restrict__ h, float* __restrict__ hn,
                         float* __restrict__ hn_out) {
  const int stride = gridDim.x * blockDim.x;
  for (int i = blockIdx.x * blockDim.x + threadIdx.x; i < Bn * DEC; i += stride) {
    const int b = i >> 9, d = i & (DEC - 1);
    const size_t base = (size_t)b * (3 * DEC);
    const float r = 1.0f / (1.0f + expf(-(gx[base + d] + gh[base + d])));
    const float z = 1.0f / (1.0f + expf(-(gx[base + DEC + d] + gh[base + DEC + d])));
    const float nn = tanhf(gx[base + 2 * DEC + d] + r * gh[base + 2 * DEC + d]);
    const float hv = (1.0f - z) * nn + z * h[i];
    hn[i] = hv; hn_out[i] = hv;
  }
}

__global__ __launch_bounds__(256)
void row_max_sum(const float* __restrict__ logits, float* __restrict__ norms) {
  const int b = blockIdx.x, tid = threadIdx.x;
  __shared__ float red[256];
  const float* row = logits + (size_t)b * V;
  float mx = -3.0e38f;
  for (int v = tid; v < V; v += 256) mx = fmaxf(mx, row[v]);
  red[tid] = mx; __syncthreads();
  for (int o = 128; o > 0; o >>= 1) { if (tid < o) red[tid] = fmaxf(red[tid], red[tid + o]); __syncthreads(); }
  mx = red[0]; __syncthreads();
  float sum = 0.0f;
  for (int v = tid; v < V; v += 256) sum += expf(row[v] - mx);
  red[tid] = sum; __syncthreads();
  for (int o = 128; o > 0; o >>= 1) { if (tid < o) red[tid] += red[tid + o]; __syncthreads(); }
  if (tid == 0) { norms[2 * b] = mx; norms[2 * b + 1] = red[0]; }
}

__global__ void finalize(const float* __restrict__ logits, const float* __restrict__ norms,
                         const float* __restrict__ pg, float* __restrict__ out) {
  const int stride = gridDim.x * blockDim.x;
  const int n = Bn * V;
  for (int i = blockIdx.x * blockDim.x + threadIdx.x; i < n; i += stride) {
    const int b = i / V;
    out[i] = pg[b] * expf(logits[i] - norms[2 * b]) / norms[2 * b + 1];
  }
}

__global__ void scatter_add(const int* __restrict__ src1, const float* __restrict__ ac,
                            const float* __restrict__ pg, float* __restrict__ out) {
  const int stride = gridDim.x * blockDim.x;
  for (int i = blockIdx.x * blockDim.x + threadIdx.x; i < S1 * Bn; i += stride) {
    const int b = i & (Bn - 1);
    atomicAdd(&out[(size_t)b * V + src1[i]], (1.0f - pg[b]) * ac[i]);
  }
}

// ---------------------------------------------------------------------------
// Host driver
// ---------------------------------------------------------------------------
extern "C" void kernel_launch(void* const* d_in, const int* in_sizes, int n_in,
                              void* d_out, int out_size, void* d_ws, size_t ws_size,
                              hipStream_t stream) {
  const int*   input      = (const int*)  d_in[0];
  const float* hidden     = (const float*)d_in[1];
  const int*   src1       = (const int*)  d_in[2];
  const float* enc1       = (const float*)d_in[3];
  const int*   mask1      = (const int*)  d_in[4];
  const int*   triple     = (const int*)  d_in[5];
  const float* enc2       = (const float*)d_in[6];
  const int*   mask2      = (const int*)  d_in[7];
  const float* embedding  = (const float*)d_in[8];
  const float* attn_W     = (const float*)d_in[9];
  const float* attn_b     = (const float*)d_in[10];
  const float* attn_v     = (const float*)d_in[11];
  const float* copy_W     = (const float*)d_in[12];
  const float* copy_b     = (const float*)d_in[13];
  const float* copy_v     = (const float*)d_in[14];
  const float* gru_Wih    = (const float*)d_in[15];
  const float* gru_Whh    = (const float*)d_in[16];
  const float* gru_bih    = (const float*)d_in[17];
  const float* gru_bhh    = (const float*)d_in[18];
  const float* fc_W       = (const float*)d_in[19];
  const float* fc_b       = (const float*)d_in[20];
  const float* gate_W     = (const float*)d_in[21];
  const float* gate_b     = (const float*)d_in[22];

  float* out_final = (float*)d_out;
  float* out_h     = out_final + (size_t)Bn * V;

  // --- workspace arena ---
  char* ws = (char*)d_ws;
  size_t off = 0;
  auto take = [&](size_t bytes) -> char* {
    char* p = ws + off;
    off += (bytes + 255) & ~(size_t)255;
    return p;
  };
  float*  emb        = (float*) take((size_t)Bn * EMB * 4);
  float*  hpA        = (float*) take((size_t)Bn * DEC * 4);
  float*  hpC        = (float*) take((size_t)Bn * DEC * 4);
  __bf16* attnWtTop  = (__bf16*)take((size_t)DEC * DEC * 2);
  __bf16* attnWtEnc  = (__bf16*)take((size_t)DEC * E2 * 2);
  __bf16* copyWtTop  = (__bf16*)take((size_t)DEC * DEC * 2);
  __bf16* copyWtEnc  = (__bf16*)take((size_t)DEC * E2 * 2);
  float*  scJoint    = (float*) take((size_t)(S1 + S2) * Bn * 4);
  float*  aJoint     = (float*) take((size_t)(S1 + S2) * Bn * 4);
  float*  wsum       = (float*) take((size_t)Bn * E2 * 4);
  float*  xg         = (float*) take((size_t)Bn * (E2 + EMB) * 4);
  float*  gx         = (float*) take((size_t)Bn * 3 * DEC * 4);
  float*  gh         = (float*) take((size_t)Bn * 3 * DEC * 4);
  float*  hnew       = (float*) take((size_t)Bn * DEC * 4);
  float*  state      = (float*) take((size_t)Bn * (DEC + E2) * 4);
  float*  pgen       = (float*) take((size_t)Bn * 4);
  float*  logits     = (float*) take((size_t)Bn * V * 4);
  float*  norms      = (float*) take((size_t)Bn * 2 * 4);
  float*  scCopy     = (float*) take((size_t)S1 * Bn * 4);
  float*  aCopy      = (float*) take((size_t)S1 * Bn * 4);

  // 1. weight transposes (f32 [K,N] -> bf16 [N,K])
  transpose_bf16<<<1024, 256, 0, stream>>>(attn_W,              attnWtTop, DEC, DEC);
  transpose_bf16<<<2048, 256, 0, stream>>>(attn_W + DEC * DEC,  attnWtEnc, E2,  DEC);
  transpose_bf16<<<1024, 256, 0, stream>>>(copy_W,              copyWtTop, DEC, DEC);
  transpose_bf16<<<2048, 256, 0, stream>>>(copy_W + DEC * DEC,  copyWtEnc, E2,  DEC);

  // 2. embedding gather
  embed_gather<<<Bn, EMB, 0, stream>>>(input, embedding, emb);

  // 3. hproj_attn = hidden @ W_top + attn_b   (WMMA, NT=8 -> grid.x = 4)
  wmma_gemm_bt<__bf16, 8><<<dim3(DEC / 128, Bn / 16), 32, 0, stream>>>(
      hidden, DEC, attnWtTop, DEC, attn_b, hpA, DEC, DEC);

  // 4. joint attention scores (fused WMMA + tanh + v-dot, 2 s per block)
  attn_scores<<<dim3(Bn / 16, S1 / 2), 128, 0, stream>>>(enc1, attnWtEnc, hpA, attn_v, scJoint);
  attn_scores<<<dim3(Bn / 16, S2 / 2), 128, 0, stream>>>(enc2, attnWtEnc, hpA, attn_v,
                                                         scJoint + (size_t)S1 * Bn);

  // 5. masked softmax over S1+S2
  masked_softmax<<<Bn, 256, 0, stream>>>(scJoint, aJoint, mask1, mask2, S1, S2, 0);

  // 6. weighted = einsum('bs,sbe->be')
  weighted_sum<<<dim3(E2 / 256, Bn), 256, 0, stream>>>(aJoint, enc1, enc2, wsum);

  // 7. x = concat(emb, weighted)
  concat2<<<1024, 256, 0, stream>>>(emb, EMB, wsum, E2, xg, Bn * (E2 + EMB));

  // 8/9. GRU gate GEMMs (WMMA, NT=8 -> grid.x = 12)
  wmma_gemm_bt<float, 8><<<dim3(3 * DEC / 128, Bn / 16), 32, 0, stream>>>(
      xg, E2 + EMB, gru_Wih, E2 + EMB, gru_bih, gx, 3 * DEC, E2 + EMB);
  wmma_gemm_bt<float, 8><<<dim3(3 * DEC / 128, Bn / 16), 32, 0, stream>>>(
      hidden, DEC, gru_Whh, DEC, gru_bhh, gh, 3 * DEC, DEC);

  // 10. GRU elementwise -> h_new (ws + output slot)
  gru_elem<<<256, 256, 0, stream>>>(gx, gh, hidden, hnew, out_h);

  // 11. state = concat(h_new, weighted)
  concat2<<<1024, 256, 0, stream>>>(hnew, DEC, wsum, E2, state, Bn * (DEC + E2));

  // 12. logits = state @ fc_W^T + fc_b  (WMMA; 3125 tiles = 5 * 625 -> NT=5)
  wmma_gemm_bt<float, 5><<<dim3(V / 80, Bn / 16), 32, 0, stream>>>(
      state, DEC + E2, fc_W, DEC + E2, fc_b, logits, V, DEC + E2);

  // 13. p_gen
  gate_kernel<<<Bn, 256, 0, stream>>>(wsum, emb, gate_W, gate_b, pgen);

  // 14/15. vocab softmax scaled by p_gen -> final_dist
  row_max_sum<<<Bn, 256, 0, stream>>>(logits, norms);
  finalize<<<4096, 256, 0, stream>>>(logits, norms, pgen, out_final);

  // 16. copy attention: hproj_copy = h_new @ copyW_top + copy_b
  wmma_gemm_bt<__bf16, 8><<<dim3(DEC / 128, Bn / 16), 32, 0, stream>>>(
      hnew, DEC, copyWtTop, DEC, copy_b, hpC, DEC, DEC);

  // 17. copy scores over encoder 1
  attn_scores<<<dim3(Bn / 16, S1 / 2), 128, 0, stream>>>(enc1, copyWtEnc, hpC, copy_v, scCopy);

  // 18. masked softmax with mask1*triple_mask
  masked_softmax<<<Bn, 256, 0, stream>>>(scCopy, aCopy, mask1, triple, S1, S1, 1);

  // 19. scatter-add copy probabilities into vocab
  scatter_add<<<256, 256, 0, stream>>>(src1, aCopy, pgen, out_final);

  (void)in_sizes; (void)n_in; (void)out_size; (void)ws_size;
}
